// LPGGRU_87101936763172
// MI455X (gfx1250) — compile-verified, matrix operand order checked by reading, not run on hardware
//
#include <hip/hip_runtime.h>
#include <hip/hip_bf16.h>
#include <cstdint>

// GRU reverse scan, MI455X (gfx1250, wave32, WMMA).
// B=1024 T=512 D=256 H=256.  Fused per-step GEMM: [x_t | h] (16x512 bf16)
// times packed W (512x1024 bf16, fragment-ordered in d_ws), fp32 accum,
// fp32 gate math, h carried in LDS.  One block = 16 batch rows, all T steps.

typedef __attribute__((ext_vector_type(16))) __bf16 v16bf;
typedef __attribute__((ext_vector_type(8)))  float  v8f;

#define B_   1024
#define T_   512
#define D_   256
#define H_   256
#define KDIM 512      // D + H
#define NCOL 1024     // [r | z | xi_n | hi_n] each 256 wide
#define NT_  64       // N tiles of 16
#define KT_  16       // K tiles of 32
#define BT_  16       // batch rows per block

// ---------------------------------------------------------------------------
// Phase 0: pack Wi/Wh (fp32) into bf16 WMMA B-fragments in d_ws.
// Logical Wpack[K=512][N=1024]:
//   N in [0,512)    : K<256 -> Wi[K][N],       K>=256 -> Wh[K-256][N]   (fused r,z)
//   N in [512,768)  : K<256 -> Wi[K][512+(N-512)], else 0               (xi_n)
//   N in [768,1024) : K>=256 -> Wh[K-256][512+(N-768)], else 0          (hi_n)
// Fragment layout per 32x16 tile (tile = nt*16+kt, 512 bf16, lane-major):
//   lane l: n = l&15 (column), g = l>>4 ; halfword hw: v=hw>>1, o=hw&1
//   Kloc = (v<4 ? 2v : 16+2(v-4)) + o + 8g   (transpose of ISA A-layout)
// ---------------------------------------------------------------------------
__global__ __launch_bounds__(256) void gru_pack_w(const float* __restrict__ Wi,
                                                  const float* __restrict__ Wh,
                                                  __bf16* __restrict__ wfrag) {
    unsigned idx = blockIdx.x * 256u + threadIdx.x;      // 524288 total
    int tile   = idx >> 9;          // 0..1023
    int nt     = tile >> 4;
    int kt     = tile & 15;
    int within = idx & 511;
    int l      = within >> 4;
    int hw     = within & 15;
    int g  = l >> 4, n16 = l & 15;
    int v  = hw >> 1, o  = hw & 1;
    int kloc = (v < 4 ? 2 * v : 16 + 2 * (v - 4)) + o + g * 8;
    int K = kt * 32 + kloc;
    int N = nt * 16 + n16;

    float val = 0.0f;
    if (N < 512) {
        val = (K < 256) ? Wi[K * 768 + N] : Wh[(K - 256) * 768 + N];
    } else if (N < 768) {
        if (K < 256) val = Wi[K * 768 + 512 + (N - 512)];
    } else {
        if (K >= 256) val = Wh[(K - 256) * 768 + 512 + (N - 768)];
    }
    wfrag[idx] = (__bf16)val;
}

// ---------------------------------------------------------------------------
// Phase 1: persistent per-batch-tile reverse scan.
// ---------------------------------------------------------------------------
union AFrag { unsigned int u[8]; v16bf v; };
union BFrag { uint4 q[2];        v16bf v; };

// B-fragment global load: tile (nt,kt), 32B per lane (two b128).
#define LOADB(dst, kt_, s_)                                                    \
    do {                                                                       \
        int nt_ = wid + 8 * (s_);                                              \
        const uint4* bp_ =                                                     \
            (const uint4*)(wf + ((size_t)((nt_) * KT_ + (kt_)) << 9)) +        \
            (lane << 1);                                                       \
        (dst).q[0] = bp_[0];                                                   \
        (dst).q[1] = bp_[1];                                                   \
    } while (0)

// A-fragment LDS gather (coalesces to 2x ds_load_b128).
#define LOADA(dst, kt_)                                                        \
    do {                                                                       \
        _Pragma("unroll") for (int v_ = 0; v_ < 8; ++v_) {                     \
            int kb_ = (v_ < 4 ? 2 * v_ : 16 + 2 * (v_ - 4)) + g8;              \
            (dst).u[v_] =                                                      \
                *(const unsigned int*)&sA[m16 * KDIM + (kt_) * 32 + kb_];      \
        }                                                                      \
    } while (0)

__global__ __launch_bounds__(256) void gru_scan(const float* __restrict__ h0,
                                                const float* __restrict__ x,
                                                const int*   __restrict__ done,
                                                const float* __restrict__ bh,
                                                const __bf16* __restrict__ wfrag,
                                                float* __restrict__ out_h,
                                                float* __restrict__ out_ys) {
    __shared__ __bf16 sA[BT_ * KDIM];     // [x_t | h] bf16, 16 KB
    __shared__ float  sG[BT_ * NCOL];     // gate pre-activations, 64 KB
    __shared__ float  sH[BT_ * H_];       // fp32 hidden carry, 16 KB
    __shared__ float  sBh[3 * H_];        // bias, 3 KB
    __shared__ int    sDone[BT_];

    const int tid  = threadIdx.x;
    const int wid  = tid >> 5;
    const int lane = tid & 31;
    const int m16  = lane & 15;           // M (or N) index within tile
    const int g8   = (lane >> 4) * 8;     // K-half / M-half selector * 8
    const int b0   = blockIdx.x * BT_;

    // init carry + bias
    for (int e = tid; e < BT_ * H_; e += 256) {
        int m = e >> 8, j = e & 255;
        sH[e] = h0[(size_t)(b0 + m) * H_ + j];
    }
    for (int e = tid; e < 3 * H_; e += 256) sBh[e] = bh[e];

#pragma unroll 1
    for (int t = T_ - 1; t >= 0; --t) {
        // Defeat LICM on the weight-fragment loads WITHOUT breaking
        // address-space inference: wfrag itself stays a plain (global-
        // inferred) kernel arg; only a zero VGPR offset is made opaque per
        // iteration.  (Making the pointer itself opaque degraded the loads
        // to flat_load_b128, which also tick DScnt and fight the LDS pipe.)
        unsigned toff = 0;
        asm volatile("" : "+v"(toff));
        const __bf16* wf = wfrag + toff;

        if (tid < BT_) sDone[tid] = done[(size_t)(b0 + tid) * T_ + t];
        __syncthreads();   // sDone ready; prev-step sH writes visible

        // ---- build A = [x_t | h_reset] in bf16 ----
        for (int e = tid; e < BT_ * KDIM; e += 256) {
            int m = e >> 9, k = e & 511;
            float val;
            if (k < D_) {
                val = x[((size_t)(b0 + m) * T_ + t) * D_ + k];
            } else {
                int j = k - D_;
                float h = sH[m * H_ + j];
                if (sDone[m]) { h = 0.0f; sH[m * H_ + j] = 0.0f; }
                val = h;
            }
            sA[e] = (__bf16)val;
        }
        __syncthreads();

        // ---- fused WMMA GEMM: G[16x1024] = A[16x512] * Wpack[512x1024] ----
        // 96 (kt,s) pairs, software-pipelined depth 3: B-fragment loads for
        // pair i+3 are issued before the WMMA of pair i (4 rotating register
        // sets), A fragments double-buffered per kt.  This turns the
        // per-WMMA "s_wait_loadcnt 0" full stalls into partial waits with
        // three loads in flight over the v_wmma stream.
        {
            v8f acc[8];
#pragma unroll
            for (int s = 0; s < 8; ++s) acc[s] = (v8f){0,0,0,0,0,0,0,0};

            AFrag a[2];
            BFrag b[4];
            LOADA(a[0], 0);
            LOADB(b[0], 0, 0);
            LOADB(b[1], 0, 1);
            LOADB(b[2], 0, 2);

#pragma unroll
            for (int i = 0; i < 96; ++i) {
                const int kt  = i / 6;
                const int pos = i % 6;
                // slots per kt: kt<8 -> {0..5}; kt>=8 -> {0,1,2,3,6,7}
                const int s   = (kt < 8) ? pos : (pos < 4 ? pos : pos + 2);

                if (pos == 0 && kt + 1 < KT_) LOADA(a[(kt + 1) & 1], kt + 1);
                if (i + 3 < 96) {
                    const int j   = i + 3;
                    const int jkt = j / 6;
                    const int jp  = j % 6;
                    const int js  = (jkt < 8) ? jp : (jp < 4 ? jp : jp + 2);
                    LOADB(b[j & 3], jkt, js);
                }
                acc[s] = __builtin_amdgcn_wmma_f32_16x16x32_bf16(
                    false, a[kt & 1].v, false, b[i & 3].v, (short)0, acc[s],
                    false, false);
            }

            // C layout: VGPR r -> M = r + 8*(lane>>4), N = lane&15
#pragma unroll
            for (int s = 0; s < 8; ++s) {
                int nt = wid + 8 * s;
#pragma unroll
                for (int r = 0; r < 8; ++r) {
                    sG[(r + g8) * NCOL + nt * 16 + m16] = acc[s][r];
                }
            }
        }
        __syncthreads();

        // ---- gate math (fp32), update carry, stream ys ----
        {
            const int j = tid;                       // 256 threads = 256 cols
            // prefetch next step's x rows while the ALUs chew on gates
            if (t > 0 && tid < BT_) {
                __builtin_prefetch(
                    &x[((size_t)(b0 + tid) * T_ + (t - 1)) * D_], 0, 3);
            }
            for (int row = 0; row < BT_; ++row) {
                const float* gr = &sG[row * NCOL];
                float rr  = gr[j]         + sBh[j];
                float zz  = gr[256 + j]   + sBh[256 + j];
                float xin = gr[512 + j];
                float hin = gr[768 + j]   + sBh[512 + j];
                float rv  = 1.0f / (1.0f + __expf(-rr));
                float zv  = 1.0f / (1.0f + __expf(-zz));
                float nv  = tanhf(xin + rv * hin);
                float hp  = sH[row * H_ + j];
                float hn  = (1.0f - zv) * nv + zv * hp;
                sH[row * H_ + j] = hn;
                out_ys[((size_t)(b0 + row) * T_ + t) * H_ + j] = hn;
            }
        }
        __syncthreads();
    }

    // ---- h_final ----
    for (int e = tid; e < BT_ * H_; e += 256) {
        int m = e >> 8, j = e & 255;
        out_h[(size_t)(b0 + m) * H_ + j] = sH[e];
    }
}

// ---------------------------------------------------------------------------
extern "C" void kernel_launch(void* const* d_in, const int* in_sizes, int n_in,
                              void* d_out, int out_size, void* d_ws, size_t ws_size,
                              hipStream_t stream) {
    const float* h0   = (const float*)d_in[0];   // gru_state [B,H]
    const float* x    = (const float*)d_in[1];   // [B,T,D]
    const int*   done = (const int*)  d_in[2];   // [B,T,1]
    const float* Wi   = (const float*)d_in[3];   // [D,3H]
    const float* Wh   = (const float*)d_in[4];   // [H,3H]
    const float* bh   = (const float*)d_in[5];   // [3H]

    __bf16* wfrag  = (__bf16*)d_ws;              // 1 MB packed fragments
    float*  out_h  = (float*)d_out;              // h_final [B,H]
    float*  out_ys = (float*)d_out + (size_t)B_ * H_;  // ys [B,T,H]

    // Phase 0: pack weights (524288 bf16 elements)
    hipLaunchKernelGGL(gru_pack_w, dim3(2048), dim3(256), 0, stream, Wi, Wh, wfrag);
    // Phase 1: 64 independent batch-tile recurrences
    hipLaunchKernelGGL(gru_scan, dim3(B_ / BT_), dim3(256), 0, stream,
                       h0, x, done, bh, wfrag, out_h, out_ys);
}
